// MOEBatched_7456063225887
// MI455X (gfx1250) — matrix operand order repaired and checked by reading, not compiled, not run on hardware
//
#include <hip/hip_runtime.h>

// ---------------- problem constants (match reference) ----------------
#define B_  4
#define S_  2048
#define D_  1024
#define I_  512
#define E_  16
#define K_  4
#define T_  512                 // tokens per expert chunk per batch = S*K/E
#define NR  (E_ * B_ * T_)      // 32768 dispatch rows

typedef __bf16 bf16;
typedef __attribute__((ext_vector_type(16))) __bf16 v16bf;
typedef __attribute__((ext_vector_type(8)))  __bf16 v8bf;
typedef __attribute__((ext_vector_type(8)))  float  v8f;

union AFrag { v16bf v; v8bf h[2]; };

// ---- CDNA5 async global->LDS copy (ASYNCcnt path), inline asm (toolchain-portable) ----
// CDNA5 ISA: each lane copies 16B from its 64-bit global address to its LDS address.
// LDS byte address = low 32 bits of the generic pointer (ISA §10.2: LDS aperture addressing).
__device__ __forceinline__ void async_load_b128(const void* g, void* lds) {
  unsigned loff = (unsigned)(unsigned long long)(uintptr_t)lds;
  unsigned long long ga = (unsigned long long)(uintptr_t)g;
  asm volatile("global_load_async_to_lds_b128 %0, %1, off"
               :
               : "v"(loff), "v"(ga)
               : "memory");
}
__device__ __forceinline__ void wait_asynccnt0() {
  asm volatile("s_wait_asynccnt 0x0" ::: "memory");
}

// ---------------- 1) f32 -> bf16 conversion ----------------
__global__ void cvt_bf16_kernel(const float* __restrict__ in, bf16* __restrict__ out, int n) {
  int i = blockIdx.x * blockDim.x + threadIdx.x;
  int stride = gridDim.x * blockDim.x;
  for (; i < n; i += stride) out[i] = (bf16)in[i];
}

// ---------------- 2) router: logits -> softmax -> top-4 ----------------
__global__ __launch_bounds__(256) void router_kernel(const float* __restrict__ x,
                                                     const float* __restrict__ rw,
                                                     float* __restrict__ ts,   // [B][S*K]
                                                     int*   __restrict__ sel)  // [B][S*K]
{
  __shared__ float part[E_][16];
  __shared__ float sc[E_];
  const int tid = threadIdx.x;
  const int b = blockIdx.x / S_, s = blockIdx.x % S_;
  const float* xr = x + ((size_t)b * S_ + s) * D_;
  const int e = tid >> 4, sub = tid & 15;
  const float* wr = rw + (size_t)e * D_;
  float acc = 0.f;
  #pragma unroll 4
  for (int u = 0; u < D_ / 16; ++u) { int d = sub + 16 * u; acc += xr[d] * wr[d]; }
  part[e][sub] = acc;
  __syncthreads();
  if (tid < E_) {
    float v = 0.f;
    #pragma unroll
    for (int k = 0; k < 16; ++k) v += part[tid][k];
    sc[tid] = v;
  }
  __syncthreads();
  if (tid == 0) {
    float m = sc[0];
    for (int i = 1; i < E_; ++i) m = fmaxf(m, sc[i]);
    float pr[E_]; float sum = 0.f;
    for (int i = 0; i < E_; ++i) { pr[i] = __expf(sc[i] - m); sum += pr[i]; }
    const float inv = 1.f / sum;
    bool used[E_];
    for (int i = 0; i < E_; ++i) used[i] = false;
    const size_t base = ((size_t)b * S_ + s) * K_;
    for (int k = 0; k < K_; ++k) {
      int best = 0; float bv = -1.f;
      for (int i = 0; i < E_; ++i)
        if (!used[i] && pr[i] > bv) { bv = pr[i]; best = i; }  // '>' keeps lowest index on ties (JAX)
      used[best] = true;
      ts[base + k]  = bv * inv;
      sel[base + k] = best;
    }
  }
}

// ---------------- 3) stable counting sort per batch ----------------
__global__ __launch_bounds__(256) void sort_kernel(const int* __restrict__ sel,        // [B][S*K]
                                                   int* __restrict__ perm_tok,         // [B][S*K] sorted-pos -> token
                                                   int* __restrict__ inv_pos)          // [B][S*K] assignment -> sorted-pos
{
  __shared__ unsigned cnt[256][E_];
  __shared__ unsigned tot[E_];
  __shared__ unsigned basee[E_];
  const int b = blockIdx.x, t = threadIdx.x;
  const int* selb = sel + (size_t)b * (S_ * K_);
  const int j0 = t * 32;                              // 8192 / 256 = 32 per thread, contiguous -> stable
  unsigned c[E_];
  #pragma unroll
  for (int e = 0; e < E_; ++e) c[e] = 0;
  int vloc[32];
  for (int jj = 0; jj < 32; ++jj) { vloc[jj] = selb[j0 + jj]; c[vloc[jj]]++; }
  #pragma unroll
  for (int e = 0; e < E_; ++e) cnt[t][e] = c[e];
  __syncthreads();
  if (t < E_) {                                       // per-expert exclusive scan over threads
    unsigned run = 0;
    for (int tt = 0; tt < 256; ++tt) { unsigned tmp = cnt[tt][t]; cnt[tt][t] = run; run += tmp; }
    tot[t] = run;
  }
  __syncthreads();
  if (t == 0) {
    unsigned r = 0;
    for (int e = 0; e < E_; ++e) { unsigned tmp = tot[e]; basee[e] = r; r += tmp; }
  }
  __syncthreads();
  unsigned offs[E_];
  #pragma unroll
  for (int e = 0; e < E_; ++e) offs[e] = basee[e] + cnt[t][e];
  int* ptk = perm_tok + (size_t)b * (S_ * K_);
  int* inv = inv_pos  + (size_t)b * (S_ * K_);
  for (int jj = 0; jj < 32; ++jj) {
    const int j = j0 + jj, e = vloc[jj];
    const unsigned p = offs[e]++;
    ptk[p] = j >> 2;     // source token = j / K
    inv[j] = (int)p;
  }
}

// ---------------- 4) gather tokens into dispatch order (bf16) ----------------
__global__ void gather_kernel(const float* __restrict__ x, const int* __restrict__ perm_tok,
                              bf16* __restrict__ Xd) {
  const int r = blockIdx.x;             // 0..NR-1 : r = e*2048 + b*512 + tin
  const int e = r >> 11;
  const int rem = r & 2047;
  const int b = rem >> 9;
  const int tin = rem & 511;
  const int p = e * T_ + tin;           // sorted position within batch b
  const int tok = perm_tok[(size_t)b * (S_ * K_) + p];
  const float* src = x + ((size_t)b * S_ + tok) * D_;
  bf16* dst = Xd + (size_t)r * D_;
  for (int d = threadIdx.x; d < D_; d += blockDim.x) dst[d] = (bf16)src[d];
}

// ---------------- 5) fused W1/W3 GEMM + SiLU*gate ----------------
// H[n,i] = silu(X[n,:]·w1[e,i,:] + b1) * (X[n,:]·w3[e,i,:] + b3)
// grid = (I/64, NR/128); block = 256 (8 waves); wave -> 16 rows x 64 cols x {w1,w3}
// W tiles staged with async global->LDS, double buffered; one barrier per K-step.
__global__ __launch_bounds__(256) void gemm13_kernel(const bf16* __restrict__ Xd,
                                                     const bf16* __restrict__ w1b,
                                                     const bf16* __restrict__ w3b,
                                                     const float* __restrict__ b1,
                                                     const float* __restrict__ b3,
                                                     bf16* __restrict__ H) {
  __shared__ alignas(16) bf16 sW1[2][64][40];   // 32 used + pad; row stride 80B (16B aligned)
  __shared__ alignas(16) bf16 sW3[2][64][40];
  const int tid = threadIdx.x;
  const int lane = tid & 31, wave = tid >> 5;
  const int e = blockIdx.y >> 4;                       // 2048 rows per expert / 128 rows per WG
  const int row0 = blockIdx.y * 128 + wave * 16;
  const int c0 = blockIdx.x * 64;
  const bf16* W1 = w1b + (size_t)e * I_ * D_;
  const bf16* W3 = w3b + (size_t)e * I_ * D_;
  const int lrow = tid >> 2, lq = tid & 3;             // 64 rows x 4 x 16B chunks
  const bf16* g1 = W1 + (size_t)(c0 + lrow) * D_ + lq * 8;
  const bf16* g3 = W3 + (size_t)(c0 + lrow) * D_ + lq * 8;
  const int arow = row0 + (lane & 15);
  const bf16* Ab = Xd + (size_t)arow * D_ + ((lane >> 4) * 16);
  const int koff = (lane >> 4) * 16;

  // prologue: fill buffer 0
  async_load_b128(g1, &sW1[0][lrow][lq * 8]);
  async_load_b128(g3, &sW3[0][lrow][lq * 8]);

  v8f acc1[4] = {}; v8f acc3[4] = {};
  for (int kk = 0; kk < D_; kk += 32) {
    const int buf = (kk >> 5) & 1;
    wait_asynccnt0();                  // this wave's fill of `buf` has landed in LDS
    __syncthreads();                   // all waves' fills done; all reads of buf^1 done
    if (kk + 32 < D_) {                // overlap: fill the other buffer during compute
      async_load_b128(g1 + kk + 32, &sW1[buf ^ 1][lrow][lq * 8]);
      async_load_b128(g3 + kk + 32, &sW3[buf ^ 1][lrow][lq * 8]);
      __builtin_prefetch(Ab + kk + 32, 0, 3);
    }
    AFrag a;                                           // 16x32 bf16 A: lanes<16 K0-15, lanes>=16 K16-31
    a.h[0] = *(const v8bf*)(Ab + kk);
    a.h[1] = *(const v8bf*)(Ab + kk + 8);
    #pragma unroll
    for (int t4 = 0; t4 < 4; ++t4) {
      const int il = t4 * 16 + (lane & 15);
      AFrag bA, bB;                                    // 32x16 B: lanes<16 K0-15, lanes>=16 K16-31
      bA.h[0] = *(const v8bf*)&sW1[buf][il][koff];
      bA.h[1] = *(const v8bf*)&sW1[buf][il][koff + 8];
      bB.h[0] = *(const v8bf*)&sW3[buf][il][koff];
      bB.h[1] = *(const v8bf*)&sW3[buf][il][koff + 8];
      acc1[t4] = __builtin_amdgcn_wmma_f32_16x16x32_bf16(false, a.v, false, bA.v, (short)0, acc1[t4], false, false);
      acc3[t4] = __builtin_amdgcn_wmma_f32_16x16x32_bf16(false, a.v, false, bB.v, (short)0, acc3[t4], false, false);
    }
  }
  // epilogue: C layout -> VGPR r: row = r + (lane>=16 ? 8:0), col = lane&15
  const int nbase = row0 + ((lane >> 4) * 8);
  const int cb = c0 + (lane & 15);
  #pragma unroll
  for (int t4 = 0; t4 < 4; ++t4) {
    const int i = cb + t4 * 16;
    const float bb1 = b1[e * I_ + i];
    const float bb3 = b3[e * I_ + i];
    #pragma unroll
    for (int r = 0; r < 8; ++r) {
      const int n = nbase + r;
      const float h1 = acc1[t4][r] + bb1;
      const float h3 = acc3[t4][r] + bb3;
      const float sig = 1.f / (1.f + __expf(-h1));
      H[(size_t)n * I_ + i] = (bf16)(h1 * sig * h3);
    }
  }
}

// ---------------- 6) W2 GEMM: OUT[n,d] = H[n,:]·w2[e,d,:] + b2[d] ----------------
// grid = (D/64, NR/128); block = 256; async double-buffered W tile
__global__ __launch_bounds__(256) void gemm2_kernel(const bf16* __restrict__ H,
                                                    const bf16* __restrict__ w2b,
                                                    const float* __restrict__ b2,
                                                    float* __restrict__ OUT) {
  __shared__ alignas(16) bf16 sW[2][64][40];
  const int tid = threadIdx.x;
  const int lane = tid & 31, wave = tid >> 5;
  const int e = blockIdx.y >> 4;
  const int row0 = blockIdx.y * 128 + wave * 16;
  const int c0 = blockIdx.x * 64;
  const bf16* W2 = w2b + (size_t)e * D_ * I_;
  const int lrow = tid >> 2, lq = tid & 3;
  const bf16* g = W2 + (size_t)(c0 + lrow) * I_ + lq * 8;
  const int arow = row0 + (lane & 15);
  const bf16* Ab = H + (size_t)arow * I_ + ((lane >> 4) * 16);
  const int koff = (lane >> 4) * 16;

  async_load_b128(g, &sW[0][lrow][lq * 8]);

  v8f acc[4] = {};
  for (int kk = 0; kk < I_; kk += 32) {
    const int buf = (kk >> 5) & 1;
    wait_asynccnt0();
    __syncthreads();
    if (kk + 32 < I_) {
      async_load_b128(g + kk + 32, &sW[buf ^ 1][lrow][lq * 8]);
      __builtin_prefetch(Ab + kk + 32, 0, 3);
    }
    AFrag a;
    a.h[0] = *(const v8bf*)(Ab + kk);
    a.h[1] = *(const v8bf*)(Ab + kk + 8);
    #pragma unroll
    for (int t4 = 0; t4 < 4; ++t4) {
      const int il = t4 * 16 + (lane & 15);
      AFrag bA;
      bA.h[0] = *(const v8bf*)&sW[buf][il][koff];
      bA.h[1] = *(const v8bf*)&sW[buf][il][koff + 8];
      acc[t4] = __builtin_amdgcn_wmma_f32_16x16x32_bf16(false, a.v, false, bA.v, (short)0, acc[t4], false, false);
    }
  }
  const int nbase = row0 + ((lane >> 4) * 8);
  const int cb = c0 + (lane & 15);
  #pragma unroll
  for (int t4 = 0; t4 < 4; ++t4) {
    const int d = cb + t4 * 16;
    const float bb = b2[e * D_ + d];
    #pragma unroll
    for (int r = 0; r < 8; ++r) {
      const int n = nbase + r;
      OUT[(size_t)n * D_ + d] = acc[t4][r] + bb;
    }
  }
}

// ---------------- 7) combine: weighted un-permute, no atomics ----------------
__global__ __launch_bounds__(256) void combine_kernel(const float* __restrict__ OUT,
                                                      const float* __restrict__ ts,
                                                      const int* __restrict__ inv_pos,
                                                      float* __restrict__ out) {
  const int b = blockIdx.x / S_, s = blockIdx.x % S_;
  float acc[4] = {0.f, 0.f, 0.f, 0.f};
  #pragma unroll
  for (int k = 0; k < K_; ++k) {
    const int j = s * K_ + k;
    const int p = inv_pos[(size_t)b * (S_ * K_) + j];
    const float w = ts[(size_t)b * (S_ * K_) + j];
    const int e = p >> 9, tin = p & 511;
    const size_t row = (size_t)e * (B_ * T_) + (size_t)b * T_ + tin;
    const float* orow = OUT + row * D_;
    #pragma unroll
    for (int u = 0; u < 4; ++u) acc[u] += w * orow[threadIdx.x + u * 256];
  }
  float* dst = out + ((size_t)b * S_ + s) * D_;
  #pragma unroll
  for (int u = 0; u < 4; ++u) dst[threadIdx.x + u * 256] = acc[u];
}

// ---------------- host side ----------------
extern "C" void kernel_launch(void* const* d_in, const int* in_sizes, int n_in,
                              void* d_out, int out_size, void* d_ws, size_t ws_size,
                              hipStream_t stream) {
  (void)in_sizes; (void)n_in; (void)out_size; (void)ws_size;
  const float* x    = (const float*)d_in[0];
  const float* r_w  = (const float*)d_in[1];
  const float* w1_w = (const float*)d_in[2];
  const float* w1_b = (const float*)d_in[3];
  const float* w2_w = (const float*)d_in[4];
  const float* w2_b = (const float*)d_in[5];
  const float* w3_w = (const float*)d_in[6];
  const float* w3_b = (const float*)d_in[7];
  float* out = (float*)d_out;

  // carve workspace (256B aligned)
  char* base = (char*)d_ws;
  size_t off = 0;
  auto carve = [&](size_t bytes) -> char* {
    size_t a = (off + 255) & ~(size_t)255;
    off = a + bytes;
    return base + a;
  };
  const size_t WN = (size_t)E_ * I_ * D_;              // 8,388,608 elems per weight tensor
  bf16*  w1b      = (bf16*)carve(WN * sizeof(bf16));
  bf16*  w3b      = (bf16*)carve(WN * sizeof(bf16));
  bf16*  w2b      = (bf16*)carve(WN * sizeof(bf16));
  bf16*  Xd       = (bf16*)carve((size_t)NR * D_ * sizeof(bf16));
  bf16*  Hbuf     = (bf16*)carve((size_t)NR * I_ * sizeof(bf16));
  float* OUT      = (float*)carve((size_t)NR * D_ * sizeof(float));
  float* ts       = (float*)carve((size_t)B_ * S_ * K_ * sizeof(float));
  int*   sel      = (int*)carve((size_t)B_ * S_ * K_ * sizeof(int));
  int*   perm_tok = (int*)carve((size_t)B_ * S_ * K_ * sizeof(int));
  int*   inv_pos  = (int*)carve((size_t)B_ * S_ * K_ * sizeof(int));

  // 1) weight conversion
  cvt_bf16_kernel<<<4096, 256, 0, stream>>>(w1_w, w1b, (int)WN);
  cvt_bf16_kernel<<<4096, 256, 0, stream>>>(w3_w, w3b, (int)WN);
  cvt_bf16_kernel<<<4096, 256, 0, stream>>>(w2_w, w2b, (int)WN);
  // 2) router
  router_kernel<<<B_ * S_, 256, 0, stream>>>(x, r_w, ts, sel);
  // 3) stable dispatch sort
  sort_kernel<<<B_, 256, 0, stream>>>(sel, perm_tok, inv_pos);
  // 4) gather + convert
  gather_kernel<<<NR, 256, 0, stream>>>(x, perm_tok, Xd);
  // 5) fused W1/W3 + SiLU*gate
  gemm13_kernel<<<dim3(I_ / 64, NR / 128), 256, 0, stream>>>(Xd, w1b, w3b, w1_b, w3_b, Hbuf);
  // 6) W2
  gemm2_kernel<<<dim3(D_ / 64, NR / 128), 256, 0, stream>>>(Hbuf, w2b, w2_b, OUT);
  // 7) combine
  combine_kernel<<<B_ * S_, 256, 0, stream>>>(OUT, ts, inv_pos, out);
}
